// MemoryDecoder_64982855189091
// MI455X (gfx1250) — matrix operand order repaired and checked by reading, not compiled, not run on hardware
//
#include <hip/hip_runtime.h>
#include <hip/hip_bf16.h>

// Problem constants (match reference)
#define T_STEPS 16
#define BATCH   32
#define HID     1024
#define EMB     512
#define L_SRC   400
#define L_SENT  24
#define L_WORD  240

typedef __attribute__((ext_vector_type(16))) __bf16 v16bf;
typedef __attribute__((ext_vector_type(8)))  float  v8f;

__device__ __forceinline__ unsigned short f2bf(float x) {
  unsigned int u = __float_as_uint(x);
  unsigned int r = ((u >> 16) & 1u) + 0x7fffu;   // round-to-nearest-even
  return (unsigned short)((u + r) >> 16);
}
__device__ __forceinline__ float bf2f(unsigned short s) {
  return __uint_as_float(((unsigned int)s) << 16);
}
__device__ __forceinline__ float sigf(float x) { return 1.0f / (1.0f + __expf(-x)); }

// ---------------------------------------------------------------------------
// WMMA GEMM: C[M,N] = (acc? C:0) + A[M,K](bf16,row-major) * W[N,K](bf16)^T
// M multiple of 16, N multiple of 16, K multiple of 32 (fast path: mult of 64).
// 8 waves per block; wave w owns output tile (blockIdx.y, blockIdx.x*8+w).
// Two independent accumulator chains (split-K) so back-to-back v_wmma ops
// have no RAW dependency on each other and can pipeline.
// If Cbf != null the result is stored as bf16 (used for Kp), else f32.
// ---------------------------------------------------------------------------
#define WPB 8
__global__ void gemm_bf16(const unsigned short* __restrict__ A,
                          const unsigned short* __restrict__ Bw,
                          float* __restrict__ Cf,
                          unsigned short* __restrict__ Cbf,
                          int M, int N, int K, int accumulate)
{
  const int wave  = threadIdx.x >> 5;
  const int lane  = threadIdx.x & 31;
  const int ntile = blockIdx.x * WPB + wave;
  const int mtile = blockIdx.y;
  if (ntile * 16 >= N) return;
  const int n0   = ntile * 16;
  const int m0   = mtile * 16;
  const int half = lane >> 4;
  const int l16  = lane & 15;

  v8f acc0 = {}, acc1 = {};
  if (accumulate) {
#pragma unroll
    for (int r = 0; r < 8; ++r)
      acc0[r] = Cf[(size_t)(m0 + half * 8 + r) * N + n0 + l16];
  }

  // A-tile: this lane supplies row m0+l16; per ISA 16-bit A layout,
  // VGPR r holds K = (r>>2)*16 + half*8 + (r&3)*2 (+1 in high half).
  // B-tile: B[k][n] = W[n][k]; lane supplies column n0+l16, VGPR r holds
  // K = (r>>2)*16 + half*8 + (r&3)*2 (+1) -> one u32 per VGPR from row of W.
  const unsigned short* arow = A  + (size_t)(m0 + l16) * K;
  const unsigned short* brow = Bw + (size_t)(n0 + l16) * K;

  const int Kmain = K & ~63;
  int k0 = 0;
  for (; k0 < Kmain; k0 += 64) {
    union { unsigned int u[8]; v16bf v; } ar0, br0, ar1, br1;
#pragma unroll
    for (int r = 0; r < 8; ++r) {
      const int kao = ((r >> 2) << 4) + (half << 3) + ((r & 3) << 1);
      const int kbo = ((r >> 2) << 4) + (half << 3) + ((r & 3) << 1);
      ar0.u[r] = *(const unsigned int*)(arow + k0 + kao);
      br0.u[r] = *(const unsigned int*)(brow + k0 + kbo);
      ar1.u[r] = *(const unsigned int*)(arow + k0 + 32 + kao);
      br1.u[r] = *(const unsigned int*)(brow + k0 + 32 + kbo);
    }
    acc0 = __builtin_amdgcn_wmma_f32_16x16x32_bf16(false, ar0.v, false, br0.v,
                                                   (short)0, acc0, false, false);
    acc1 = __builtin_amdgcn_wmma_f32_16x16x32_bf16(false, ar1.v, false, br1.v,
                                                   (short)0, acc1, false, false);
  }
  for (; k0 < K; k0 += 32) {          // tail (K multiple of 32)
    union { unsigned int u[8]; v16bf v; } ar, br;
#pragma unroll
    for (int r = 0; r < 8; ++r) {
      ar.u[r] = *(const unsigned int*)(arow + k0 + ((r >> 2) << 4) + (half << 3) + ((r & 3) << 1));
      br.u[r] = *(const unsigned int*)(brow + k0 + ((r >> 2) << 4) + (half << 3) + ((r & 3) << 1));
    }
    acc0 = __builtin_amdgcn_wmma_f32_16x16x32_bf16(false, ar.v, false, br.v,
                                                   (short)0, acc0, false, false);
  }

#pragma unroll
  for (int r = 0; r < 8; ++r) {
    const float res = acc0[r] + acc1[r];
    const size_t o = (size_t)(m0 + half * 8 + r) * N + n0 + l16;
    if (Cbf) Cbf[o] = f2bf(res);
    else     Cf[o]  = res;
  }
}

// ---------------------------------------------------------------------------
// one-time converters
// ---------------------------------------------------------------------------
__global__ void cvt_bf(const float* __restrict__ in, unsigned short* __restrict__ out,
                       size_t n) {
  for (size_t i = (size_t)blockIdx.x * blockDim.x + threadIdx.x; i < n;
       i += (size_t)gridDim.x * blockDim.x)
    out[i] = f2bf(in[i]);
}
// out[c*R + r] = in[r*C + c]  (transpose, for Wq / Wk_* used non-transposed)
__global__ void cvt_bf_T(const float* __restrict__ in, unsigned short* __restrict__ out,
                         int R, int C) {
  const size_t n = (size_t)R * C;
  for (size_t i = (size_t)blockIdx.x * blockDim.x + threadIdx.x; i < n;
       i += (size_t)gridDim.x * blockDim.x) {
    int r = (int)(i / C), c = (int)(i % C);
    out[(size_t)c * R + r] = f2bf(in[i]);
  }
}
__global__ void zero_f32(float* p, int n) {
  int i = blockIdx.x * blockDim.x + threadIdx.x;
  if (i < n) p[i] = 0.0f;
}

// ---------------------------------------------------------------------------
// per-step kernels
// ---------------------------------------------------------------------------
// xcat = bf16([emb[tgt[b]], feed[b]]); hprev_bf = bf16(h)
__global__ void embed_concat(const int* __restrict__ tgt_t,
                             const float* __restrict__ table,
                             const float* __restrict__ feed,
                             const float* __restrict__ h,
                             unsigned short* __restrict__ xcat,
                             unsigned short* __restrict__ hprev_bf)
{
  const int b = blockIdx.x;
  const int tok = tgt_t[b];
  const float* er = table + (size_t)tok * EMB;
  unsigned short* xr = xcat + (size_t)b * (EMB + HID);
  for (int e = threadIdx.x; e < EMB; e += 256) xr[e] = f2bf(er[e]);
  for (int d = threadIdx.x; d < HID; d += 256) {
    xr[EMB + d]                     = f2bf(feed[(size_t)b * HID + d]);
    hprev_bf[(size_t)b * HID + d]   = f2bf(h[(size_t)b * HID + d]);
  }
}

__global__ void lstm_pw(const float* __restrict__ gates,
                        const float* __restrict__ b_ih, const float* __restrict__ b_hh,
                        float* __restrict__ c, float* __restrict__ h,
                        unsigned short* __restrict__ hq_bf,
                        unsigned short* __restrict__ cat_qa,
                        unsigned short* __restrict__ cat_g,
                        unsigned short* __restrict__ cat_out)
{
  int idx = blockIdx.x * blockDim.x + threadIdx.x;
  if (idx >= BATCH * HID) return;
  int b = idx / HID, d = idx % HID;
  const float* g = gates + (size_t)b * 4 * HID;
  float i_ = g[d]           + b_ih[d]           + b_hh[d];
  float f_ = g[HID + d]     + b_ih[HID + d]     + b_hh[HID + d];
  float g_ = g[2*HID + d]   + b_ih[2*HID + d]   + b_hh[2*HID + d];
  float o_ = g[3*HID + d]   + b_ih[3*HID + d]   + b_hh[3*HID + d];
  float cn = sigf(f_) * c[idx] + sigf(i_) * tanhf(g_);
  float hn = sigf(o_) * tanhf(cn);
  c[idx] = cn;
  h[idx] = hn;
  unsigned short hb = f2bf(hn);
  hq_bf[idx] = hb;
  cat_qa[(size_t)b * 3 * HID + 2 * HID + d] = hb;
  cat_g [(size_t)b * 3 * HID + 2 * HID + d] = hb;
  cat_out[(size_t)b * 2 * HID + HID + d]    = hb;
}

// score[b,l] = v . tanh(Kp[l,b,:] + qW[b,:] (+ cov[b,l]*wc)) ; -1e9 if masked
__global__ void att_score(const unsigned short* __restrict__ Kp,   // [L,B,H] bf16
                          const float* __restrict__ qW,            // [B,H]
                          const float* __restrict__ cov,           // [B,L] or null
                          const float* __restrict__ wc,            // [H] or null
                          const float* __restrict__ v,             // [H]
                          const int* __restrict__ lens,            // [B]
                          float* __restrict__ score,               // [B,L]
                          int L)
{
  const int l = blockIdx.x, b = blockIdx.y;
  __shared__ float red[256];
  if (l >= lens[b]) {
    if (threadIdx.x == 0) score[(size_t)b * L + l] = -1e9f;
    return;
  }
  const float cv = cov ? cov[(size_t)b * L + l] : 0.0f;
  const unsigned short* kp = Kp + ((size_t)l * BATCH + b) * HID;
  const float* q = qW + (size_t)b * HID;
  float p = 0.0f;
  for (int hh = threadIdx.x; hh < HID; hh += 256) {
    float pre = bf2f(kp[hh]) + q[hh];
    if (wc) pre += cv * wc[hh];
    p += tanhf(pre) * v[hh];
  }
  red[threadIdx.x] = p;
  __syncthreads();
  for (int s = 128; s > 0; s >>= 1) {
    if (threadIdx.x < s) red[threadIdx.x] += red[threadIdx.x + s];
    __syncthreads();
  }
  if (threadIdx.x == 0) score[(size_t)b * L + l] = red[0];
}

// softmax over L (in-place: scores -> probs) then ctx[b,:] = sum_l a_l*bank[l,b,:]
__global__ void att_sm_ctx(float* __restrict__ score,              // [B,L]
                           const float* __restrict__ bank,         // [L,B,H] f32
                           float* __restrict__ ctx,                // [B,H] f32 or null
                           unsigned short* __restrict__ ctx_bf,    // concat slice base
                           int ctx_ld, int L)
{
  const int b = blockIdx.x;
  __shared__ float a_sh[L_SRC];                                    // max L = 400
  __shared__ float red[256];
  float* sc = score + (size_t)b * L;

  float mx = -3.0e38f;
  for (int l = threadIdx.x; l < L; l += 256) mx = fmaxf(mx, sc[l]);
  red[threadIdx.x] = mx; __syncthreads();
  for (int s = 128; s > 0; s >>= 1) {
    if (threadIdx.x < s) red[threadIdx.x] = fmaxf(red[threadIdx.x], red[threadIdx.x + s]);
    __syncthreads();
  }
  mx = red[0]; __syncthreads();

  float sum = 0.0f;
  for (int l = threadIdx.x; l < L; l += 256) {
    float e = __expf(sc[l] - mx);
    a_sh[l] = e; sum += e;
  }
  red[threadIdx.x] = sum; __syncthreads();
  for (int s = 128; s > 0; s >>= 1) {
    if (threadIdx.x < s) red[threadIdx.x] += red[threadIdx.x + s];
    __syncthreads();
  }
  const float inv = 1.0f / red[0];
  __syncthreads();
  for (int l = threadIdx.x; l < L; l += 256) {
    float a = a_sh[l] * inv;
    a_sh[l] = a;
    sc[l] = a;                                  // probs back out (for coverage)
  }
  __syncthreads();

  float a0 = 0, a1 = 0, a2 = 0, a3 = 0;
  const int h0 = threadIdx.x;
  for (int l = 0; l < L; ++l) {
    const float a = a_sh[l];
    const float* bl = bank + ((size_t)l * BATCH + b) * HID;
    a0 += a * bl[h0];
    a1 += a * bl[h0 + 256];
    a2 += a * bl[h0 + 512];
    a3 += a * bl[h0 + 768];
  }
  if (ctx) {
    float* cb = ctx + (size_t)b * HID;
    cb[h0] = a0; cb[h0 + 256] = a1; cb[h0 + 512] = a2; cb[h0 + 768] = a3;
  }
  unsigned short* cf = ctx_bf + (size_t)b * ctx_ld;
  cf[h0] = f2bf(a0); cf[h0 + 256] = f2bf(a1);
  cf[h0 + 512] = f2bf(a2); cf[h0 + 768] = f2bf(a3);
}

// qa_ctx = sig(lin)*sctx + (1-sig(lin))*wctx ; bf16 into cat_g[:,H:2H]
__global__ void pw_gqa(const float* __restrict__ lin, const float* __restrict__ sctx,
                       const float* __restrict__ wctx, float* __restrict__ qactx,
                       unsigned short* __restrict__ cat_g)
{
  int idx = blockIdx.x * blockDim.x + threadIdx.x;
  if (idx >= BATCH * HID) return;
  int b = idx / HID, d = idx % HID;
  float g  = sigf(lin[idx]);
  float qa = g * sctx[idx] + (1.0f - g) * wctx[idx];
  qactx[idx] = qa;
  cat_g[(size_t)b * 3 * HID + HID + d] = f2bf(qa);
}

// src_context = sig(lin)*pctx + (1-sig(lin))*qactx ; bf16 into cat_out[:,0:H]
__global__ void pw_g2(const float* __restrict__ lin, const float* __restrict__ pctx,
                      const float* __restrict__ qactx, unsigned short* __restrict__ cat_out)
{
  int idx = blockIdx.x * blockDim.x + threadIdx.x;
  if (idx >= BATCH * HID) return;
  int b = idx / HID, d = idx % HID;
  float g  = sigf(lin[idx]);
  float sc = g * pctx[idx] + (1.0f - g) * qactx[idx];
  cat_out[(size_t)b * 2 * HID + d] = f2bf(sc);
}

__global__ void pw_out(const float* __restrict__ lin, float* __restrict__ out) {
  int idx = blockIdx.x * blockDim.x + threadIdx.x;
  if (idx >= BATCH * HID) return;
  out[idx] = tanhf(lin[idx]);
}

// cov = (t==0) ? attn : 0.85*cov + 0.15*exp(-attn)
__global__ void cov_update(float* __restrict__ cov, const float* __restrict__ attn,
                           int n, int t)
{
  int i = blockIdx.x * blockDim.x + threadIdx.x;
  if (i >= n) return;
  cov[i] = (t == 0) ? attn[i] : 0.85f * cov[i] + 0.15f * __expf(-attn[i]);
}

// ---------------------------------------------------------------------------
extern "C" void kernel_launch(void* const* d_in, const int* in_sizes, int n_in,
                              void* d_out, int out_size, void* d_ws, size_t ws_size,
                              hipStream_t stream)
{
  (void)in_sizes; (void)n_in; (void)out_size; (void)ws_size;
  const int*   tgt        = (const int*)  d_in[0];
  const float* src_bank   = (const float*)d_in[1];
  const int*   src_len    = (const int*)  d_in[2];
  const float* sent_bank  = (const float*)d_in[3];
  const int*   sent_len   = (const int*)  d_in[4];
  const float* word_bank  = (const float*)d_in[5];
  const int*   word_len   = (const int*)  d_in[6];
  const float* input_feed = (const float*)d_in[7];
  const float* h0         = (const float*)d_in[8];
  const float* c0         = (const float*)d_in[9];
  const float* embedding  = (const float*)d_in[10];
  const float* W_ih       = (const float*)d_in[11];
  const float* W_hh       = (const float*)d_in[12];
  const float* b_ih       = (const float*)d_in[13];
  const float* b_hh       = (const float*)d_in[14];
  const float* Wq         = (const float*)d_in[15];
  const float* Wk_src     = (const float*)d_in[16];
  const float* Wk_sent    = (const float*)d_in[17];
  const float* Wk_word    = (const float*)d_in[18];
  const float* v_src      = (const float*)d_in[19];
  const float* v_sent     = (const float*)d_in[20];
  const float* v_word     = (const float*)d_in[21];
  const float* wc_src     = (const float*)d_in[22];
  const float* wc_word    = (const float*)d_in[23];
  const float* Wg_qa      = (const float*)d_in[24];
  const float* Wg         = (const float*)d_in[25];
  const float* W_out      = (const float*)d_in[26];
  float* out = (float*)d_out;

  // ----- workspace carve (bump allocator, 256B aligned) -----
  size_t off = 0;
  auto alloc = [&](size_t bytes) -> void* {
    off = (off + 255) & ~(size_t)255;
    void* p = (char*)d_ws + off;
    off += bytes;
    return p;
  };
  auto aU16 = [&](size_t n) { return (unsigned short*)alloc(n * 2); };
  auto aF32 = [&](size_t n) { return (float*)alloc(n * 4); };

  unsigned short* wih_bf    = aU16((size_t)4 * HID * (EMB + HID));
  unsigned short* whh_bf    = aU16((size_t)4 * HID * HID);
  unsigned short* wqT_bf    = aU16((size_t)HID * HID);
  unsigned short* wksrcT    = aU16((size_t)HID * HID);
  unsigned short* wksentT   = aU16((size_t)HID * HID);
  unsigned short* wkwordT   = aU16((size_t)HID * HID);
  unsigned short* wgqa_bf   = aU16((size_t)HID * 3 * HID);
  unsigned short* wg_bf     = aU16((size_t)HID * 3 * HID);
  unsigned short* wout_bf   = aU16((size_t)HID * 2 * HID);
  unsigned short* srcb_bf   = aU16((size_t)L_SRC  * BATCH * HID);
  unsigned short* sentb_bf  = aU16((size_t)L_SENT * BATCH * HID);
  unsigned short* wordb_bf  = aU16((size_t)L_WORD * BATCH * HID);
  unsigned short* kp_src    = aU16((size_t)L_SRC  * BATCH * HID);
  unsigned short* kp_sent   = aU16((size_t)L_SENT * BATCH * HID);
  unsigned short* kp_word   = aU16((size_t)L_WORD * BATCH * HID);
  unsigned short* xcat      = aU16((size_t)BATCH * (EMB + HID));
  unsigned short* hprev_bf  = aU16((size_t)BATCH * HID);
  unsigned short* hq_bf     = aU16((size_t)BATCH * HID);
  unsigned short* cat_qa    = aU16((size_t)BATCH * 3 * HID);
  unsigned short* cat_g     = aU16((size_t)BATCH * 3 * HID);
  unsigned short* cat_out   = aU16((size_t)BATCH * 2 * HID);
  float* gates  = aF32((size_t)BATCH * 4 * HID);
  float* hbuf   = aF32((size_t)BATCH * HID);
  float* cbuf   = aF32((size_t)BATCH * HID);
  float* qWbuf  = aF32((size_t)BATCH * HID);
  float* sctx   = aF32((size_t)BATCH * HID);
  float* wctx   = aF32((size_t)BATCH * HID);
  float* pctx   = aF32((size_t)BATCH * HID);
  float* qactx  = aF32((size_t)BATCH * HID);
  float* lin    = aF32((size_t)BATCH * HID);
  float* sc_src  = aF32((size_t)BATCH * L_SRC);
  float* sc_sent = aF32((size_t)BATCH * L_SENT);
  float* sc_word = aF32((size_t)BATCH * L_WORD);
  float* cov    = aF32((size_t)BATCH * L_SRC);
  float* cov2   = aF32((size_t)BATCH * L_WORD);

  auto gemm = [&](const unsigned short* A, const unsigned short* Bw,
                  float* Cf, unsigned short* Cbf, int M, int N, int K, int acc) {
    dim3 g((N + 16 * WPB - 1) / (16 * WPB), M / 16);
    gemm_bf16<<<g, dim3(32 * WPB), 0, stream>>>(A, Bw, Cf, Cbf, M, N, K, acc);
  };

  // ----- one-time: bf16 weight/bank conversion (deterministic per call) -----
  const int CVG = 2048;
  cvt_bf<<<CVG, 256, 0, stream>>>(W_ih,  wih_bf,  (size_t)4 * HID * (EMB + HID));
  cvt_bf<<<CVG, 256, 0, stream>>>(W_hh,  whh_bf,  (size_t)4 * HID * HID);
  cvt_bf<<<CVG, 256, 0, stream>>>(Wg_qa, wgqa_bf, (size_t)HID * 3 * HID);
  cvt_bf<<<CVG, 256, 0, stream>>>(Wg,    wg_bf,   (size_t)HID * 3 * HID);
  cvt_bf<<<CVG, 256, 0, stream>>>(W_out, wout_bf, (size_t)HID * 2 * HID);
  cvt_bf<<<CVG, 256, 0, stream>>>(src_bank,  srcb_bf,  (size_t)L_SRC  * BATCH * HID);
  cvt_bf<<<CVG, 256, 0, stream>>>(sent_bank, sentb_bf, (size_t)L_SENT * BATCH * HID);
  cvt_bf<<<CVG, 256, 0, stream>>>(word_bank, wordb_bf, (size_t)L_WORD * BATCH * HID);
  cvt_bf_T<<<CVG, 256, 0, stream>>>(Wq,      wqT_bf,  HID, HID);
  cvt_bf_T<<<CVG, 256, 0, stream>>>(Wk_src,  wksrcT,  HID, HID);
  cvt_bf_T<<<CVG, 256, 0, stream>>>(Wk_sent, wksentT, HID, HID);
  cvt_bf_T<<<CVG, 256, 0, stream>>>(Wk_word, wkwordT, HID, HID);

  // ----- hoisted key projections: Kp = bank @ Wk (WMMA, bf16 out) -----
  gemm(srcb_bf,  wksrcT,  nullptr, kp_src,  L_SRC  * BATCH, HID, HID, 0);
  gemm(sentb_bf, wksentT, nullptr, kp_sent, L_SENT * BATCH, HID, HID, 0);
  gemm(wordb_bf, wkwordT, nullptr, kp_word, L_WORD * BATCH, HID, HID, 0);

  // ----- state init -----
  hipMemcpyAsync(hbuf, h0, (size_t)BATCH * HID * 4, hipMemcpyDeviceToDevice, stream);
  hipMemcpyAsync(cbuf, c0, (size_t)BATCH * HID * 4, hipMemcpyDeviceToDevice, stream);
  zero_f32<<<(BATCH * L_SRC  + 255) / 256, 256, 0, stream>>>(cov,  BATCH * L_SRC);
  zero_f32<<<(BATCH * L_WORD + 255) / 256, 256, 0, stream>>>(cov2, BATCH * L_WORD);

  const int PW_G = (BATCH * HID + 255) / 256;

  // ----- decoder timestep loop -----
  for (int t = 0; t < T_STEPS; ++t) {
    const float* feed = (t == 0) ? input_feed : out + (size_t)(t - 1) * BATCH * HID;

    embed_concat<<<BATCH, 256, 0, stream>>>(tgt + t * BATCH, embedding, feed,
                                            hbuf, xcat, hprev_bf);
    // gates = [e,feed] @ W_ih^T + h @ W_hh^T
    gemm(xcat,     wih_bf, gates, nullptr, BATCH, 4 * HID, EMB + HID, 0);
    gemm(hprev_bf, whh_bf, gates, nullptr, BATCH, 4 * HID, HID,       1);
    lstm_pw<<<PW_G, 256, 0, stream>>>(gates, b_ih, b_hh, cbuf, hbuf,
                                      hq_bf, cat_qa, cat_g, cat_out);
    // qW = h_n @ Wq
    gemm(hq_bf, wqT_bf, qWbuf, nullptr, BATCH, HID, HID, 0);

    // attention scores
    att_score<<<dim3(L_SRC,  BATCH), 256, 0, stream>>>(kp_src,  qWbuf, cov,     wc_src,
                                                       v_src,  src_len,  sc_src,  L_SRC);
    att_score<<<dim3(L_SENT, BATCH), 256, 0, stream>>>(kp_sent, qWbuf, nullptr, nullptr,
                                                       v_sent, sent_len, sc_sent, L_SENT);
    att_score<<<dim3(L_WORD, BATCH), 256, 0, stream>>>(kp_word, qWbuf, cov2,    wc_word,
                                                       v_word, word_len, sc_word, L_WORD);
    // softmax + context (bf16 copies land directly in concat buffers)
    att_sm_ctx<<<BATCH, 256, 0, stream>>>(sc_src,  src_bank,  pctx, cat_g,        3 * HID, L_SRC);
    att_sm_ctx<<<BATCH, 256, 0, stream>>>(sc_sent, sent_bank, sctx, cat_qa,       3 * HID, L_SENT);
    att_sm_ctx<<<BATCH, 256, 0, stream>>>(sc_word, word_bank, wctx, cat_qa + HID, 3 * HID, L_WORD);

    // qa gate
    gemm(cat_qa, wgqa_bf, lin, nullptr, BATCH, HID, 3 * HID, 0);
    pw_gqa<<<PW_G, 256, 0, stream>>>(lin, sctx, wctx, qactx, cat_g);
    // source/qa gate
    gemm(cat_g, wg_bf, lin, nullptr, BATCH, HID, 3 * HID, 0);
    pw_g2<<<PW_G, 256, 0, stream>>>(lin, pctx, qactx, cat_out);
    // output projection
    gemm(cat_out, wout_bf, lin, nullptr, BATCH, HID, 2 * HID, 0);
    pw_out<<<PW_G, 256, 0, stream>>>(lin, out + (size_t)t * BATCH * HID);

    // coverage EMA updates
    cov_update<<<(BATCH * L_SRC  + 255) / 256, 256, 0, stream>>>(cov,  sc_src,  BATCH * L_SRC,  t);
    cov_update<<<(BATCH * L_WORD + 255) / 256, 256, 0, stream>>>(cov2, sc_word, BATCH * L_WORD, t);
  }
}
